// GNNEncoder_35734127903105
// MI455X (gfx1250) — matrix operand order repaired and checked by reading, not compiled, not run on hardware
//
#include <hip/hip_runtime.h>
#include <hip/hip_bf16.h>

typedef __attribute__((ext_vector_type(16))) _Float16 v16h;
typedef __attribute__((ext_vector_type(8)))  _Float16 v8h;
typedef __attribute__((ext_vector_type(4)))  _Float16 v4h;
typedef __attribute__((ext_vector_type(8)))  float    v8f;

#define NNODES 50000
#define NGRAPH 8
#define NPG    6250
#define NH     4
#define DD     128
#define HD     512   /* NH*DD */
#define NEDGE  400000

// ---------------------------------------------------------------- utilities

__global__ void k_fill_f32(float* __restrict__ p, float v, long long n) {
  long long i = (long long)blockIdx.x * blockDim.x + threadIdx.x;
  long long s = (long long)gridDim.x * blockDim.x;
  for (; i < n; i += s) p[i] = v;
}

// W [K,N] f32 row-major  ->  WT [N,K] f16 (so WMMA B-fragments are contiguous)
__global__ void k_w_transpose_f16(const float* __restrict__ W, _Float16* __restrict__ WT,
                                  int K, int N) {
  long long t = (long long)blockIdx.x * blockDim.x + threadIdx.x;
  if (t >= (long long)K * N) return;
  const int n = (int)(t % N);
  const int k = (int)(t / N);
  WT[(long long)n * K + k] = (_Float16)W[t];   // coalesced read of W
}

__device__ __forceinline__ void atomicMaxFloat(float* addr, float value) {
  // classic sign-split trick: correct for mixed-sign maxima
  if (value >= 0.f) atomicMax((int*)addr, __float_as_int(value));
  else              atomicMin((unsigned int*)addr, __float_as_uint(value));
}

__device__ __forceinline__ float waveSum(float v) {
#pragma unroll
  for (int o = 16; o > 0; o >>= 1) v += __shfl_xor(v, o, 32);
  return v;
}

// ------------------------------------------------------- embedding gather

__global__ void k_gather_embed(const int* __restrict__ ids, const float* __restrict__ emb,
                               float* __restrict__ feat, _Float16* __restrict__ feat_h) {
  long long t = (long long)blockIdx.x * blockDim.x + threadIdx.x;
  const long long total = (long long)NNODES * (DD / 4);
  if (t >= total) return;
  long long n = t / (DD / 4);
  int c = (int)(t % (DD / 4)) * 4;
  const float4 v = *(const float4*)(emb + (long long)ids[n] * DD + c);
  *(float4*)(feat + n * DD + c) = v;
  v4h h; h[0] = (_Float16)v.x; h[1] = (_Float16)v.y; h[2] = (_Float16)v.z; h[3] = (_Float16)v.w;
  *(v4h*)(feat_h + n * DD + c) = h;
}

// ------------------------------------------------------------ WMMA GEMM
// C[M,Ncols] = A[M,128] * BT^T + bias ; BT is [Ncols,128] f16 (pre-transposed W)
// one wave computes a 16x64 strip: 4 N-tiles register-blocked, K unrolled as
// 4 x v_wmma_f32_16x16x32_f16 each -> 16 WMMAs/wave, all operand loads b128.

template <bool F16OUT, bool LEAKY>
__global__ __launch_bounds__(128)
void k_wmma_gemm(const _Float16* __restrict__ A, const _Float16* __restrict__ BT,
                 const float* __restrict__ bias, void* __restrict__ Cout,
                 int M, int Ncols) {
  const int wave = threadIdx.x >> 5;
  const int lane = threadIdx.x & 31;
  const int l15  = lane & 15;
  const int nblk = Ncols >> 6;                               // 64-wide N blocks
  const long long job = (long long)blockIdx.x * (blockDim.x >> 5) + wave;
  const long long totalJobs = (long long)(M >> 4) * nblk;
  if (job >= totalJobs) return;
  const int tm   = (int)(job / nblk);
  const int tnb  = (int)(job % nblk);
  const int row  = tm * 16 + l15;
  const int cb   = tnb * 64;                                 // first output col
  const int hiA  = (lane & 16) ? 8 : 0;                      // unused; A loads use hiB too
  const int hiB  = (lane & 16) ? 16 : 0;                     // B half-group K offset

  const _Float16* __restrict__ arow = A + (long long)row * 128;
  const _Float16* __restrict__ b0 = BT + (long long)(cb + 0 * 16 + l15) * 128 + hiB;
  const _Float16* __restrict__ b1 = BT + (long long)(cb + 1 * 16 + l15) * 128 + hiB;
  const _Float16* __restrict__ b2 = BT + (long long)(cb + 2 * 16 + l15) * 128 + hiB;
  const _Float16* __restrict__ b3 = BT + (long long)(cb + 3 * 16 + l15) * 128 + hiB;

  v8f acc0 = {}, acc1 = {}, acc2 = {}, acc3 = {};
#pragma unroll
  for (int k0 = 0; k0 < 128; k0 += 32) {
    const v8h alo = *(const v8h*)(arow + k0 + hiB);
    const v8h ahi = *(const v8h*)(arow + k0 + hiB + 8);
    const v16h a = __builtin_shufflevector(alo, ahi, 0, 1, 2, 3, 4, 5, 6, 7,
                                           8, 9, 10, 11, 12, 13, 14, 15);
    const v16h f0 = *(const v16h*)(b0 + k0);   // 32B aligned -> 2x b128
    const v16h f1 = *(const v16h*)(b1 + k0);
    const v16h f2 = *(const v16h*)(b2 + k0);
    const v16h f3 = *(const v16h*)(b3 + k0);
    acc0 = __builtin_amdgcn_wmma_f32_16x16x32_f16(false, a, false, f0, (short)0, acc0, false, false);
    acc1 = __builtin_amdgcn_wmma_f32_16x16x32_f16(false, a, false, f1, (short)0, acc1, false, false);
    acc2 = __builtin_amdgcn_wmma_f32_16x16x32_f16(false, a, false, f2, (short)0, acc2, false, false);
    acc3 = __builtin_amdgcn_wmma_f32_16x16x32_f16(false, a, false, f3, (short)0, acc3, false, false);
  }

  const int rbase = tm * 16 + ((lane & 16) ? 8 : 0);
#pragma unroll
  for (int j = 0; j < 4; ++j) {
    const v8f acc = (j == 0) ? acc0 : (j == 1) ? acc1 : (j == 2) ? acc2 : acc3;
    const int col = cb + j * 16 + l15;
    const float bv = bias ? bias[col] : 0.f;
#pragma unroll
    for (int v = 0; v < 8; ++v) {
      float x = acc[v] + bv;
      if (LEAKY) x = (x >= 0.f) ? x : 0.01f * x;
      if (F16OUT) ((_Float16*)Cout)[(long long)(rbase + v) * Ncols + col] = (_Float16)x;
      else        ((float*)Cout)[(long long)(rbase + v) * Ncols + col] = x;
    }
  }
}

// ------------------------------------------------- GATv2 edge attention

// wave per (edge, head): logits[e,h] = sum_d attn[h,d]*leaky(hs[src,h,d]+hd[dst,h,d],0.2)
__global__ __launch_bounds__(256)
void k_edge_logits(const int* __restrict__ src, const int* __restrict__ dst,
                   const _Float16* __restrict__ hs, const _Float16* __restrict__ hd,
                   const float* __restrict__ attn,
                   float* __restrict__ logits, float* __restrict__ mmax) {
  const int lane = threadIdx.x & 31;
  const long long w = (long long)blockIdx.x * (blockDim.x >> 5) + (threadIdx.x >> 5);
  if (w >= (long long)NEDGE * NH) return;
  const int e = (int)(w >> 2);
  const int h = (int)(w & 3);
  const int s = src[e], d = dst[e];
  const int off = h * DD + lane * 4;
  const v4h a = *(const v4h*)(hs + (long long)s * HD + off);
  const v4h b = *(const v4h*)(hd + (long long)d * HD + off);
  const float4 at = *(const float4*)(attn + off);
  float p = 0.f, t;
  t = (float)a[0] + (float)b[0]; t = (t >= 0.f) ? t : 0.2f * t; p += at.x * t;
  t = (float)a[1] + (float)b[1]; t = (t >= 0.f) ? t : 0.2f * t; p += at.y * t;
  t = (float)a[2] + (float)b[2]; t = (t >= 0.f) ? t : 0.2f * t; p += at.z * t;
  t = (float)a[3] + (float)b[3]; t = (t >= 0.f) ? t : 0.2f * t; p += at.w * t;
  p = waveSum(p);
  if (lane == 0) {
    logits[w] = p;
    atomicMaxFloat(&mmax[(long long)d * NH + h], p);
  }
}

// thread per (edge, head): ex = exp(logit - m[dst]); denom[dst] += ex
__global__ void k_edge_softmax_num(const int* __restrict__ dst, const float* __restrict__ logits,
                                   const float* __restrict__ mmax,
                                   float* __restrict__ ex, float* __restrict__ denom) {
  long long t = (long long)blockIdx.x * blockDim.x + threadIdx.x;
  if (t >= (long long)NEDGE * NH) return;
  const int e = (int)(t >> 2);
  const int h = (int)(t & 3);
  const int d = dst[e];
  const float v = __expf(logits[t] - mmax[(long long)d * NH + h]);
  ex[t] = v;
  atomicAdd(&denom[(long long)d * NH + h], v);
}

// wave per (edge, head): rst[dst,h,:] += hs[src,h,:] * alpha
__global__ __launch_bounds__(256)
void k_edge_aggregate(const int* __restrict__ src, const int* __restrict__ dst,
                      const _Float16* __restrict__ hs, const float* __restrict__ ex,
                      const float* __restrict__ denom, float* __restrict__ rst) {
  const int lane = threadIdx.x & 31;
  const long long w = (long long)blockIdx.x * (blockDim.x >> 5) + (threadIdx.x >> 5);
  if (w >= (long long)NEDGE * NH) return;
  const int e = (int)(w >> 2);
  const int h = (int)(w & 3);
  const int s = src[e], d = dst[e];
  const float alpha = ex[w] / denom[(long long)d * NH + h];
  const int off = h * DD + lane * 4;
  const v4h a = *(const v4h*)(hs + (long long)s * HD + off);
  float* r = rst + (long long)d * HD + off;
  atomicAdd(r + 0, (float)a[0] * alpha);
  atomicAdd(r + 1, (float)a[1] * alpha);
  atomicAdd(r + 2, (float)a[2] * alpha);
  atomicAdd(r + 3, (float)a[3] * alpha);
}

// per (node,d): out1 = mean_h leaky(rst + res_lin, 0.01)
__global__ void k_gat_finalize(const float* __restrict__ rst, const _Float16* __restrict__ resl,
                               float* __restrict__ out1) {
  long long t = (long long)blockIdx.x * blockDim.x + threadIdx.x;
  if (t >= (long long)NNODES * DD) return;
  const long long n = t / DD;
  const int d = (int)(t % DD);
  float acc = 0.f;
#pragma unroll
  for (int h = 0; h < NH; ++h) {
    float x = rst[n * HD + h * DD + d] + (float)resl[n * HD + h * DD + d];
    acc += (x >= 0.f) ? x : 0.01f * x;
  }
  out1[t] = acc * 0.25f;
}

// ------------------------------------------------------------ GraphConv

__global__ void k_degrees(const int* __restrict__ src, const int* __restrict__ dst,
                          float* __restrict__ dout, float* __restrict__ din) {
  long long e = (long long)blockIdx.x * blockDim.x + threadIdx.x;
  if (e >= NEDGE) return;
  atomicAdd(&dout[src[e]], 1.f);
  atomicAdd(&din[dst[e]], 1.f);
}

__global__ void k_scale_src(const float* __restrict__ x, const float* __restrict__ dout,
                            float* __restrict__ y) {
  long long t = (long long)blockIdx.x * blockDim.x + threadIdx.x;
  if (t >= (long long)NNODES * DD) return;
  float dg = dout[t / DD]; dg = (dg < 1.f) ? 1.f : dg;
  y[t] = x[t] * rsqrtf(dg);
}

// wave per edge: agg[dst,:] += hscaled[src,:]
__global__ __launch_bounds__(256)
void k_gc_aggregate(const int* __restrict__ src, const int* __restrict__ dst,
                    const float* __restrict__ hsc, float* __restrict__ agg) {
  const int lane = threadIdx.x & 31;
  const long long w = (long long)blockIdx.x * (blockDim.x >> 5) + (threadIdx.x >> 5);
  if (w >= NEDGE) return;
  const int s = src[w], d = dst[w];
  const float4 v = *(const float4*)(hsc + (long long)s * DD + lane * 4);
  float* a = agg + (long long)d * DD + lane * 4;
  atomicAdd(a + 0, v.x);
  atomicAdd(a + 1, v.y);
  atomicAdd(a + 2, v.z);
  atomicAdd(a + 3, v.w);
}

__global__ void k_scale_dst_cvt(const float* __restrict__ agg, const float* __restrict__ din,
                                _Float16* __restrict__ agg_h) {
  long long t = (long long)blockIdx.x * blockDim.x + threadIdx.x;
  if (t >= (long long)NNODES * DD) return;
  float dg = din[t / DD]; dg = (dg < 1.f) ? 1.f : dg;
  agg_h[t] = (_Float16)(agg[t] * rsqrtf(dg));
}

// ----------------------------------------------------- pooling + readout

__global__ __launch_bounds__(128)
void k_pool_max(const float* __restrict__ x, float* __restrict__ pool) {
  const int b = blockIdx.x, d = threadIdx.x;
  float m = -1e30f;
  const float* p = x + (long long)b * NPG * DD + d;
  for (int i = 0; i < NPG; ++i) {
    if (i + 8 < NPG) __builtin_prefetch(p + (long long)(i + 8) * DD, 0, 1);
    m = fmaxf(m, p[(long long)i * DD]);
  }
  pool[b * DD + d] = m;
}

__global__ __launch_bounds__(128)
void k_final_mlp(const float* __restrict__ p0, const float* __restrict__ p1,
                 const float* __restrict__ p2, const float* __restrict__ Wmlp,
                 const float* __restrict__ bmlp, float* __restrict__ out) {
  __shared__ float red[128];
  const int b = blockIdx.x, d = threadIdx.x;
  float s = fmaxf(p0[b * DD + d], 0.f) * Wmlp[d]
          + fmaxf(p1[b * DD + d], 0.f) * Wmlp[DD + d]
          + fmaxf(p2[b * DD + d], 0.f) * Wmlp[2 * DD + d];
  red[d] = s;
  __syncthreads();
  for (int o = 64; o > 0; o >>= 1) { if (d < o) red[d] += red[d + o]; __syncthreads(); }
  if (d == 0) out[b] = red[0] + bmlp[0];
}

// ---------------------------------------------------------------- launch

extern "C" void kernel_launch(void* const* d_in, const int* in_sizes, int n_in,
                              void* d_out, int out_size, void* d_ws, size_t ws_size,
                              hipStream_t stream) {
  const int*   node_ids = (const int*)d_in[0];
  const int*   src      = (const int*)d_in[1];
  const int*   dst      = (const int*)d_in[2];
  const float* emb      = (const float*)d_in[3];
  const float* W_src    = (const float*)d_in[4];
  const float* b_src    = (const float*)d_in[5];
  const float* W_dst    = (const float*)d_in[6];
  const float* b_dst    = (const float*)d_in[7];
  const float* attn     = (const float*)d_in[8];
  const float* W_res    = (const float*)d_in[9];
  const float* b_res    = (const float*)d_in[10];
  const float* W_gc     = (const float*)d_in[11];
  const float* b_gc     = (const float*)d_in[12];
  const float* W_mlp    = (const float*)d_in[13];
  const float* b_mlp    = (const float*)d_in[14];
  float* out = (float*)d_out;

  char* wsp = (char*)d_ws;
  size_t off = 0;
  auto alloc = [&](size_t bytes) -> void* {
    void* p = wsp + off;
    off = (off + bytes + 255) & ~(size_t)255;
    return p;
  };

  float*    feat    = (float*)   alloc((size_t)NNODES * DD * 4);
  _Float16* feat_h  = (_Float16*)alloc((size_t)NNODES * DD * 2);
  _Float16* Wsrc_t  = (_Float16*)alloc((size_t)DD * HD * 2);   // [512,128] f16
  _Float16* Wdst_t  = (_Float16*)alloc((size_t)DD * HD * 2);
  _Float16* Wres_t  = (_Float16*)alloc((size_t)DD * HD * 2);
  _Float16* Wgc_t   = (_Float16*)alloc((size_t)DD * DD * 2);   // [128,128] f16
  _Float16* hs_h    = (_Float16*)alloc((size_t)NNODES * HD * 2);
  _Float16* hd_h    = (_Float16*)alloc((size_t)NNODES * HD * 2);
  _Float16* res_h   = (_Float16*)alloc((size_t)NNODES * HD * 2);
  float*    logits  = (float*)   alloc((size_t)NEDGE * NH * 4);
  float*    mmax    = (float*)   alloc((size_t)NNODES * NH * 4);
  float*    exbuf   = (float*)   alloc((size_t)NEDGE * NH * 4);
  float*    denom   = (float*)   alloc((size_t)NNODES * NH * 4);
  float*    rst     = (float*)   alloc((size_t)NNODES * HD * 4);
  float*    out1    = (float*)   alloc((size_t)NNODES * DD * 4);
  float*    hscaled = (float*)   alloc((size_t)NNODES * DD * 4);
  float*    agg     = (float*)   alloc((size_t)NNODES * DD * 4);
  _Float16* agg_h   = (_Float16*)alloc((size_t)NNODES * DD * 2);
  float*    out2    = (float*)   alloc((size_t)NNODES * DD * 4);
  float*    dout    = (float*)   alloc((size_t)NNODES * 4);
  float*    din     = (float*)   alloc((size_t)NNODES * 4);
  float*    pool0   = (float*)   alloc((size_t)NGRAPH * DD * 4);
  float*    pool1   = (float*)   alloc((size_t)NGRAPH * DD * 4);
  float*    pool2   = (float*)   alloc((size_t)NGRAPH * DD * 4);
  (void)ws_size; (void)in_sizes; (void)n_in; (void)out_size;

  // ---- initialize accumulators (fresh every call: graph-replay safe)
  k_fill_f32<<<8192, 256, 0, stream>>>(rst, 0.f, (long long)NNODES * HD);
  k_fill_f32<<<4096, 256, 0, stream>>>(agg, 0.f, (long long)NNODES * DD);
  k_fill_f32<<<512, 256, 0, stream>>>(denom, 0.f, (long long)NNODES * NH);
  k_fill_f32<<<512, 256, 0, stream>>>(mmax, -1e30f, (long long)NNODES * NH);
  k_fill_f32<<<256, 256, 0, stream>>>(dout, 0.f, NNODES);
  k_fill_f32<<<256, 256, 0, stream>>>(din, 0.f, NNODES);

  // ---- featurize + weight transpose/conversion
  k_gather_embed<<<(NNODES * (DD / 4) + 255) / 256, 256, 0, stream>>>(node_ids, emb, feat, feat_h);
  k_w_transpose_f16<<<(DD * HD + 255) / 256, 256, 0, stream>>>(W_src, Wsrc_t, DD, HD);
  k_w_transpose_f16<<<(DD * HD + 255) / 256, 256, 0, stream>>>(W_dst, Wdst_t, DD, HD);
  k_w_transpose_f16<<<(DD * HD + 255) / 256, 256, 0, stream>>>(W_res, Wres_t, DD, HD);
  k_w_transpose_f16<<<(DD * DD + 255) / 256, 256, 0, stream>>>(W_gc, Wgc_t, DD, DD);

  // ---- GATv2 linear projections: [50000,128]x[128,512] via WMMA (16x64 strips)
  {
    const int jobs = (NNODES / 16) * (HD / 64);       // 25000
    const int blocks = (jobs + 3) / 4;                // 4 waves/block
    k_wmma_gemm<true, false><<<blocks, 128, 0, stream>>>(feat_h, Wsrc_t, b_src, hs_h, NNODES, HD);
    k_wmma_gemm<true, false><<<blocks, 128, 0, stream>>>(feat_h, Wdst_t, b_dst, hd_h, NNODES, HD);
    k_wmma_gemm<true, false><<<blocks, 128, 0, stream>>>(feat_h, Wres_t, b_res, res_h, NNODES, HD);
  }

  // ---- edge softmax + aggregation
  {
    const long long ewaves = (long long)NEDGE * NH;   // 1.6M (edge,head) waves
    const int eblocks = (int)((ewaves + 7) / 8);      // 8 waves/block
    k_edge_logits<<<eblocks, 256, 0, stream>>>(src, dst, hs_h, hd_h, attn, logits, mmax);
    k_edge_softmax_num<<<(int)((ewaves + 255) / 256), 256, 0, stream>>>(dst, logits, mmax, exbuf, denom);
    k_edge_aggregate<<<eblocks, 256, 0, stream>>>(src, dst, hs_h, exbuf, denom, rst);
  }
  k_gat_finalize<<<(NNODES * DD + 255) / 256, 256, 0, stream>>>(rst, res_h, out1);

  // ---- GraphConv
  k_degrees<<<(NEDGE + 255) / 256, 256, 0, stream>>>(src, dst, dout, din);
  k_scale_src<<<(NNODES * DD + 255) / 256, 256, 0, stream>>>(out1, dout, hscaled);
  k_gc_aggregate<<<(NEDGE + 7) / 8, 256, 0, stream>>>(src, dst, hscaled, agg);
  k_scale_dst_cvt<<<(NNODES * DD + 255) / 256, 256, 0, stream>>>(agg, din, agg_h);
  {
    const int jobs = (NNODES / 16) * (DD / 64);       // 6250
    k_wmma_gemm<false, true><<<(jobs + 3) / 4, 128, 0, stream>>>(agg_h, Wgc_t, b_gc, out2, NNODES, DD);
  }

  // ---- per-graph max pools + readout MLP
  k_pool_max<<<NGRAPH, 128, 0, stream>>>(feat, pool0);
  k_pool_max<<<NGRAPH, 128, 0, stream>>>(out1, pool1);
  k_pool_max<<<NGRAPH, 128, 0, stream>>>(out2, pool2);
  k_final_mlp<<<NGRAPH, 128, 0, stream>>>(pool0, pool1, pool2, W_mlp, b_mlp, out);
}
